// NonLocalInteraction_13005160972684
// MI455X (gfx1250) — matrix operand order repaired and checked by reading, not compiled, not run on hardware
//
#include <hip/hip_runtime.h>

typedef __attribute__((ext_vector_type(16))) _Float16 v16h;
typedef __attribute__((ext_vector_type(8)))  float    v8f;

#define DEVINL static __device__ __forceinline__

constexpr int   LDF  = 132;   // f32 LDS row stride (floats)
constexpr int   LDH  = 136;   // f16 LDS row stride (halves)
constexpr int   MAXA = 96;    // static pad size
constexpr int   FD   = 128;   // feat dim = nb_features
constexpr float KEPS = 1e-4f;

constexpr size_t SMEM_BYTES =
    (size_t)(2 * MAXA * LDF) * 4 +                 // xbuf + hbuf (f32)
    (size_t)(4 * MAXA * LDH + FD * LDH) * 2 +      // a16,q16,k16,v16b + wb (f16)
    (size_t)(FD + 4 * MAXA + 32) * 4;              // ksum, diagq/k, dinv, rowred, blkred

DEVINL v8f zero8() {
  v8f z;
#pragma unroll
  for (int i = 0; i < 8; ++i) z[i] = 0.0f;
  return z;
}

DEVINL v8f wmma16(v16h a, v16h b, v8f c) {
  // D = A(16x32 f16) x B(32x16 f16) + C(16x16 f32)
  return __builtin_amdgcn_wmma_f32_16x16x32_f16(false, a, false, b, (short)0, c,
                                                false, false);
}

// A fragment 16x32 f16 at (row0,k0) from row-major LDS (stride LDH halves).
// Lane L: M = L%16; lanes 0-15 hold K = [0..8)+[16..24), lanes 16-31 shifted by 8.
DEVINL v16h load_a(const _Float16* A, int row0, int k0, int lane) {
  const int m  = lane & 15;
  const int kb = (lane < 16) ? 0 : 8;
  const _Float16* p = A + (row0 + m) * LDH + k0 + kb;
  v16h a;
#pragma unroll
  for (int j = 0; j < 8; ++j) a[j] = p[j];
#pragma unroll
  for (int j = 0; j < 8; ++j) a[8 + j] = p[16 + j];
  return a;
}

// A fragment of S^T: logical A[m][k] = S[k][m] (transposed gather from LDS).
DEVINL v16h load_at(const _Float16* S, int row0, int k0, int lane) {
  const int m  = lane & 15;
  const int kb = (lane < 16) ? 0 : 8;
  const _Float16* p = S + (k0 + kb) * LDH + row0 + m;
  v16h a;
#pragma unroll
  for (int j = 0; j < 8; ++j) a[j] = p[j * LDH];
#pragma unroll
  for (int j = 0; j < 8; ++j) a[8 + j] = p[(16 + j) * LDH];
  return a;
}

// B fragment 32x16 f16 at (k0,col0): lane L holds column N=L%16,
// lanes 0-15 K=[0..8)+[16..24), lanes 16-31 shifted by 8.
DEVINL v16h load_b(const _Float16* B, int k0, int col0, int lane) {
  const int n  = lane & 15;
  const int kb = (lane < 16) ? 0 : 8;
  const _Float16* p = B + (k0 + kb) * LDH + col0 + n;
  v16h b;
#pragma unroll
  for (int j = 0; j < 8; ++j) b[j] = p[j * LDH];
#pragma unroll
  for (int j = 0; j < 8; ++j) b[8 + j] = p[(16 + j) * LDH];
  return b;
}

// C/D 16x16 f32: lane L holds N=L%16; lanes 0-15 M=0-7, lanes 16-31 M=8-15.
DEVINL void store_c(float* C, v8f acc, int row0, int col0, int lane) {
  const int n  = lane & 15;
  const int mb = (lane < 16) ? 0 : 8;
#pragma unroll
  for (int v = 0; v < 8; ++v) C[(row0 + mb + v) * LDF + col0 + n] = acc[v];
}

DEVINL void store_ch(_Float16* C, v8f acc, int row0, int col0, int lane) {
  const int n  = lane & 15;
  const int mb = (lane < 16) ? 0 : 8;
#pragma unroll
  for (int v = 0; v < 8; ++v)
    C[(row0 + mb + v) * LDH + col0 + n] = (_Float16)acc[v];
}

// C[rows x 128] (f32, stride LDF) = A16[rows x 128] @ B16[128 x 128], K=128.
// 8 waves: wave w owns column tile w; B fragments loaded once, reused over rows.
DEVINL void gemm_rk(const _Float16* A, const _Float16* B, float* C,
                    int rowTiles, int wave, int lane) {
  const int ct = wave * 16;
  const v16h b0 = load_b(B, 0,  ct, lane);
  const v16h b1 = load_b(B, 32, ct, lane);
  const v16h b2 = load_b(B, 64, ct, lane);
  const v16h b3 = load_b(B, 96, ct, lane);
  for (int rt = 0; rt < rowTiles; ++rt) {
    v8f acc = zero8();
    acc = wmma16(load_a(A, rt * 16, 0,  lane), b0, acc);
    acc = wmma16(load_a(A, rt * 16, 32, lane), b1, acc);
    acc = wmma16(load_a(A, rt * 16, 64, lane), b2, acc);
    acc = wmma16(load_a(A, rt * 16, 96, lane), b3, acc);
    store_c(C, acc, rt * 16, ct, lane);
  }
}

// context[128 x 128] (f16) = phiK^T[128 x na] @ V[na x 128], K padded to 32*ksteps.
DEVINL void gemm_ctx(const _Float16* phiK, const _Float16* V, _Float16* C,
                     int ksteps, int wave, int lane) {
  const int ct = wave * 16;
  for (int rt = 0; rt < 8; ++rt) {
    v8f acc = zero8();
    for (int ks = 0; ks < ksteps; ++ks)
      acc = wmma16(load_at(phiK, rt * 16, ks * 32, lane),
                   load_b(V, ks * 32, ct, lane), acc);
    store_ch(C, acc, rt * 16, ct, lane);
  }
}

// Async DMA: 128x128 f16 tile (32KB) global -> LDS (stride LDH), 128b per lane,
// no VGPR round trip; tracked by ASYNCcnt and overlapped with VALU work.
DEVINL void async_load_w(_Float16* dst, const _Float16* src, int tid) {
  const unsigned lbase = (unsigned)(uintptr_t)dst;   // LDS logical offset (addr[31:0])
  const char* sbase = (const char*)src;
#pragma unroll
  for (int it = 0; it < 8; ++it) {
    const int i  = tid + it * 256;                   // 2048 x 16B chunks
    const int r  = i >> 4;
    const int c8 = (i & 15) << 3;
    const unsigned laddr = lbase + (unsigned)(r * LDH + c8) * 2u;
    const unsigned voff  = (unsigned)i << 4;
    asm volatile("global_load_async_to_lds_b128 %0, %1, %2"
                 :
                 : "v"(laddr), "v"(voff), "s"(sbase)
                 : "memory");
  }
}

DEVINL void wait_async() {
  asm volatile("s_wait_asynccnt 0x0" ::: "memory");
}

DEVINL float lswish(float x, float a, float b) {
  return a * x * (1.0f / (1.0f + __expf(-b * x)));
}

__global__ void prep_offsets_kernel(const int* __restrict__ na,
                                    int* __restrict__ offs, int G) {
  const int g = blockIdx.x * blockDim.x + threadIdx.x;
  if (g < G) {
    int s = 0;
    for (int i = 0; i < g; ++i) s += na[i];
    offs[g] = s;
  }
}

__global__ void prep_convert_kernel(const float* __restrict__ W1,
                                    const float* __restrict__ W2,
                                    const float* __restrict__ W3,
                                    const float* __restrict__ proj,
                                    _Float16* __restrict__ W1h,
                                    _Float16* __restrict__ W2h,
                                    _Float16* __restrict__ W3h,
                                    _Float16* __restrict__ projTh) {
  const int idx = blockIdx.x * blockDim.x + threadIdx.x;
  if (idx < 3 * FD * FD) {
    W1h[idx] = (_Float16)W1[idx];
    W2h[idx] = (_Float16)W2[idx];
    W3h[idx] = (_Float16)W3[idx];
  }
  if (idx < FD * FD) {                  // projT[f][m] = proj[m][f]
    const int m = idx >> 7, f = idx & 127;
    projTh[f * FD + m] = (_Float16)proj[idx];
  }
}

__global__ __launch_bounds__(256)
void performer_fused(const float* __restrict__ x,
                     const int* __restrict__ num_atoms,
                     const int* __restrict__ offs,
                     const _Float16* __restrict__ W1h, const float* __restrict__ b1,
                     const _Float16* __restrict__ W2h, const float* __restrict__ b2,
                     const _Float16* __restrict__ W3h, const float* __restrict__ b3,
                     const float* __restrict__ pa1, const float* __restrict__ pb1,
                     const float* __restrict__ pa2, const float* __restrict__ pb2,
                     const float* __restrict__ pa3, const float* __restrict__ pb3,
                     const _Float16* __restrict__ projTh,
                     float* __restrict__ out) {
  extern __shared__ char smem_raw[];
  float*    xbuf   = (float*)smem_raw;                 // [96][LDF] input (residual)
  float*    hbuf   = xbuf + MAXA * LDF;                // [96][LDF] f32 GEMM out
  _Float16* a16    = (_Float16*)(hbuf + MAXA * LDF);   // [96][LDH] activation A
  _Float16* q16    = a16  + MAXA * LDH;                // qn -> phi_q
  _Float16* k16    = q16  + MAXA * LDH;                // kn -> phi_k
  _Float16* v16b   = k16  + MAXA * LDH;                // V
  _Float16* wb     = v16b + MAXA * LDH;                // [128][LDH] W / projT / context
  float*    ksum   = (float*)(wb + FD * LDH);          // [128]
  float*    diagq  = ksum  + FD;                       // [96]
  float*    diagk  = diagq + MAXA;
  float*    dinv   = diagk + MAXA;
  float*    rowred = dinv  + MAXA;
  float*    blkred = rowred + MAXA;

  const int g    = blockIdx.x;
  const int na   = num_atoms[g];
  const int base = offs[g];
  const int tid  = threadIdx.x;
  const int wave = tid >> 5;
  const int lane = tid & 31;
  const int rowTiles = (na + 15) >> 4;
  const int kstepsN  = (na + 31) >> 5;

  const float al1 = pa1[0], be1s = pb1[0];
  const float al2 = pa2[0], be2s = pb2[0];
  const float al3 = pa3[0], be3s = pb3[0];
  const float dn    = 0.29730177875068026f;   // 128^-0.25
  const float ratio = 0.08838834764831845f;   // 128^-0.5

  // ---- load this graph's x tile ----
  for (int idx = tid; idx < na * FD; idx += 256) {
    const int r = idx >> 7, c = idx & 127;
    xbuf[r * LDF + c] = x[(size_t)(base + r) * FD + c];
  }
  __syncthreads();

  // ---- 3-head residual MLP -> qn (q16), kn (k16), V (v16b), all f16 ----
  for (int k = 0; k < 3; ++k) {
    const _Float16* Wsrc1 = W1h + (size_t)k * FD * FD;
    const _Float16* Wsrc2 = W2h + (size_t)k * FD * FD;
    const _Float16* Wsrc3 = W3h + (size_t)k * FD * FD;
    __builtin_prefetch(Wsrc2, 0, 0);
    __builtin_prefetch(Wsrc3, 0, 0);

    // layer 1: weight DMA overlapped with a16 = lswish1(x)
    async_load_w(wb, Wsrc1, tid);
    for (int idx = tid; idx < MAXA * FD; idx += 256) {
      const int r = idx >> 7, c = idx & 127;
      const float v = (r < na) ? lswish(xbuf[r * LDF + c], al1, be1s) : 0.0f;
      a16[r * LDH + c] = (_Float16)v;
    }
    wait_async();
    __syncthreads();
    gemm_rk(a16, wb, hbuf, rowTiles, wave, lane);
    __syncthreads();

    // layer 2: a16 = lswish2(h1 + b1)
    async_load_w(wb, Wsrc2, tid);
    for (int idx = tid; idx < MAXA * FD; idx += 256) {
      const int r = idx >> 7, c = idx & 127;
      const float v = (r < na)
          ? lswish(hbuf[r * LDF + c] + b1[k * FD + c], al2, be2s) : 0.0f;
      a16[r * LDH + c] = (_Float16)v;
    }
    wait_async();
    __syncthreads();
    gemm_rk(a16, wb, hbuf, rowTiles, wave, lane);
    __syncthreads();

    // layer 3: res = x + h2 + b2; a16 = lswish3(res)
    async_load_w(wb, Wsrc3, tid);
    for (int idx = tid; idx < MAXA * FD; idx += 256) {
      const int r = idx >> 7, c = idx & 127;
      const float v = (r < na)
          ? lswish(xbuf[r * LDF + c] + hbuf[r * LDF + c] + b2[k * FD + c],
                   al3, be3s) : 0.0f;
      a16[r * LDH + c] = (_Float16)v;
    }
    wait_async();
    __syncthreads();
    gemm_rk(a16, wb, hbuf, rowTiles, wave, lane);
    __syncthreads();

    // epilogue: y = x + h3 + b3 ; apply data_normalizer for Q/K
    for (int idx = tid; idx < MAXA * FD; idx += 256) {
      const int r = idx >> 7, c = idx & 127;
      float y = 0.0f;
      if (r < na) y = xbuf[r * LDF + c] + hbuf[r * LDF + c] + b3[k * FD + c];
      if (k == 0)      q16[r * LDH + c]  = (_Float16)(y * dn);
      else if (k == 1) k16[r * LDH + c]  = (_Float16)(y * dn);
      else             v16b[r * LDH + c] = (_Float16)y;
    }
    __syncthreads();
  }

  // ---- diag terms, overlapped with projT DMA into wb ----
  async_load_w(wb, projTh, tid);
  if (tid < na && tid < MAXA) {
    float sq = 0.0f, sk = 0.0f;
    for (int c = 0; c < FD; ++c) {
      const float qv = (float)q16[tid * LDH + c];
      const float kv = (float)k16[tid * LDH + c];
      sq += qv * qv; sk += kv * kv;
    }
    diagq[tid] = 0.5f * sq;
    diagk[tid] = 0.5f * sk;
  }
  wait_async();
  __syncthreads();

  // ---- dd_k = kn @ projT ; mk = per-graph max ; phi_k ----
  gemm_rk(k16, wb, hbuf, rowTiles, wave, lane);
  __syncthreads();
  if (tid < MAXA) {
    float m = -3.0e38f;
    if (tid < na)
      for (int c = 0; c < FD; ++c) m = fmaxf(m, hbuf[tid * LDF + c]);
    rowred[tid] = m;
  }
  __syncthreads();
  if (tid == 0) {
    float m = -3.0e38f;
    for (int r = 0; r < na; ++r) m = fmaxf(m, rowred[r]);
    blkred[0] = m;
  }
  __syncthreads();
  const float mk = blkred[0];
  for (int idx = tid; idx < MAXA * FD; idx += 256) {
    const int r = idx >> 7, c = idx & 127;
    float v = 0.0f;
    if (r < na)
      v = ratio * (__expf(hbuf[r * LDF + c] - diagk[r] - mk) + KEPS);
    k16[r * LDH + c] = (_Float16)v;     // k16 := phi_k
  }
  __syncthreads();
  if (tid < FD) {                        // k_sum[m] = sum_n phi_k[n][m]
    float s = 0.0f;
    for (int r = 0; r < na; ++r) s += (float)k16[r * LDH + tid];
    ksum[tid] = s;
  }
  __syncthreads();

  // ---- dd_q = qn @ projT ; per-row max ; phi_q ----
  gemm_rk(q16, wb, hbuf, rowTiles, wave, lane);
  __syncthreads();
  if (tid < MAXA) {
    float m = -3.0e38f;
    if (tid < na)
      for (int c = 0; c < FD; ++c) m = fmaxf(m, hbuf[tid * LDF + c]);
    rowred[tid] = m;
  }
  __syncthreads();
  for (int idx = tid; idx < MAXA * FD; idx += 256) {
    const int r = idx >> 7, c = idx & 127;
    float v = 0.0f;
    if (r < na)
      v = ratio * (__expf(hbuf[r * LDF + c] - diagq[r] - rowred[r]) + KEPS);
    q16[r * LDH + c] = (_Float16)v;     // q16 := phi_q
  }
  __syncthreads();
  if (tid < na && tid < MAXA) {          // d_inv = 1 / (phi_q . k_sum)
    float s = 0.0f;
    for (int c = 0; c < FD; ++c) s += (float)q16[tid * LDH + c] * ksum[c];
    dinv[tid] = 1.0f / s;
  }
  __syncthreads();

  // ---- context = phi_k^T @ V  (f16, overwrites projT staging) ----
  gemm_ctx(k16, v16b, wb, kstepsN, wave, lane);
  __syncthreads();

  // ---- out = (phi_q @ context) * d_inv ----
  gemm_rk(q16, wb, hbuf, rowTiles, wave, lane);
  __syncthreads();
  for (int idx = tid; idx < na * FD; idx += 256) {
    const int r = idx >> 7, c = idx & 127;
    out[(size_t)(base + r) * FD + c] = hbuf[r * LDF + c] * dinv[r];
  }
}

extern "C" void kernel_launch(void* const* d_in, const int* in_sizes, int n_in,
                              void* d_out, int out_size, void* d_ws, size_t ws_size,
                              hipStream_t stream) {
  const float* x    = (const float*)d_in[0];
  const int*   nat  = (const int*)d_in[1];
  const float* W1   = (const float*)d_in[2];
  const float* b1   = (const float*)d_in[3];
  const float* W2   = (const float*)d_in[4];
  const float* b2   = (const float*)d_in[5];
  const float* W3   = (const float*)d_in[6];
  const float* b3   = (const float*)d_in[7];
  const float* a1   = (const float*)d_in[8];
  const float* be1  = (const float*)d_in[9];
  const float* a2   = (const float*)d_in[10];
  const float* be2  = (const float*)d_in[11];
  const float* a3   = (const float*)d_in[12];
  const float* be3  = (const float*)d_in[13];
  const float* proj = (const float*)d_in[14];
  const int G = in_sizes[1];
  (void)n_in; (void)out_size; (void)ws_size;

  char* ws = (char*)d_ws;
  int* offs = (int*)ws;
  size_t o = ((size_t)G * sizeof(int) + 255) & ~(size_t)255;
  _Float16* W1h    = (_Float16*)(ws + o); o += (size_t)3 * FD * FD * 2;
  _Float16* W2h    = (_Float16*)(ws + o); o += (size_t)3 * FD * FD * 2;
  _Float16* W3h    = (_Float16*)(ws + o); o += (size_t)3 * FD * FD * 2;
  _Float16* projTh = (_Float16*)(ws + o); o += (size_t)FD * FD * 2;

  prep_offsets_kernel<<<(G + 255) / 256, 256, 0, stream>>>(nat, offs, G);
  prep_convert_kernel<<<(3 * FD * FD + 255) / 256, 256, 0, stream>>>(
      W1, W2, W3, proj, W1h, W2h, W3h, projTh);

  (void)hipFuncSetAttribute(reinterpret_cast<const void*>(&performer_fused),
                            hipFuncAttributeMaxDynamicSharedMemorySize,
                            (int)SMEM_BYTES);
  performer_fused<<<G, 256, SMEM_BYTES, stream>>>(
      x, nat, offs, W1h, b1, W2h, b2, W3h, b3,
      a1, be1, a2, be2, a3, be3, projTh, (float*)d_out);
}
